// GCN_69458211111263
// MI455X (gfx1250) — matrix operand order repaired and checked by reading, not compiled, not run on hardware
//
#include <hip/hip_runtime.h>
#include <hip/hip_bf16.h>

typedef __attribute__((ext_vector_type(16))) _Float16 v16h;
typedef __attribute__((ext_vector_type(8)))  float    v8f;

#define N_NODES   50000
#define N_GRAPHS  128
#define HID       128
#define OUT_DIM   2

// ---------------------------------------------------------------------------
// small helpers
// ---------------------------------------------------------------------------
__device__ __forceinline__ unsigned enc_f32(float f) {
    unsigned u = __float_as_uint(f);
    return (u & 0x80000000u) ? ~u : (u | 0x80000000u);   // monotone order-preserving
}
__device__ __forceinline__ float dec_f32(unsigned e) {
    unsigned u = (e & 0x80000000u) ? (e & 0x7FFFFFFFu) : ~e;
    return __uint_as_float(u);
}

// ---------------------------------------------------------------------------
// degree / normalization
// ---------------------------------------------------------------------------
__global__ void k_deg_init(float* deg, int n) {
    int i = blockIdx.x * blockDim.x + threadIdx.x;
    if (i < n) deg[i] = 1.0f;                 // self loop contributes 1
}
__global__ void k_deg_edges(const int* __restrict__ col, float* deg, int e) {
    int i = blockIdx.x * blockDim.x + threadIdx.x;
    if (i < e) atomicAdd(&deg[col[i]], 1.0f);
}
__global__ void k_dinv(float* deg, int n) {
    int i = blockIdx.x * blockDim.x + threadIdx.x;
    if (i < n) { float d = deg[i]; deg[i] = d > 0.0f ? rsqrtf(d) : 0.0f; }
}

// ---------------------------------------------------------------------------
// fp32 -> f16 conversion (GEMM operands)
// ---------------------------------------------------------------------------
__global__ void k_f32_to_f16(const float* __restrict__ src, _Float16* __restrict__ dst, int n) {
    int i = blockIdx.x * blockDim.x + threadIdx.x;
    if (i < n) dst[i] = (_Float16)src[i];
}

// ---------------------------------------------------------------------------
// WMMA GEMM (fused epilogue):
//   v        = x @ w.T + bias          (f16 WMMA, f32 accumulate)
//   xw[i]    = v                       (needed by edge scatter gather)
//   acc[i]   = dinv[i]^2 * v           (self-loop init of aggregation buffer)
// Each wave computes a 16x32 tile (two accumulators share one A per K-step
// -> 8 v_wmma per wave). 8 waves/block -> 32 rows x 128 cols.
// ---------------------------------------------------------------------------
__global__ void k_gemm_xw_wmma(const _Float16* __restrict__ xh,   // [n,128]
                               const _Float16* __restrict__ wh,   // [128,128]
                               const float*    __restrict__ bias, // [128]
                               const float*    __restrict__ dinv, // [n]
                               float*          __restrict__ xw,   // [n,128]
                               float*          __restrict__ acc,  // [n,128]
                               int n_rows)
{
    const int lane = threadIdx.x & 31;
    const int wave = threadIdx.x >> 5;                     // 0..7
    const int row0 = blockIdx.x * 32 + (wave >> 2) * 16;   // 16-row tile
    if (row0 >= n_rows) return;
    const int col0 = (wave & 3) * 32;                      // 32-col tile

    const int m    = lane & 15;
    int srow       = row0 + m;
    if (srow >= n_rows) srow = n_rows - 1;                 // clamp; masked on store
    const int nidx = lane & 15;
    const int kA   = (lane < 16) ? 0 : 8;                  // A 16x32 f16 layout (ISA 7.12.2)
    const int kB   = (lane < 16) ? 0 : 8;                  // B 32x16 f16 layout

    v8f acc0 = {}, acc1 = {};
    const _Float16* arow = xh + (size_t)srow * HID;
    const _Float16* b0p  = wh + (size_t)(col0 + nidx) * HID + kB;
    const _Float16* b1p  = wh + (size_t)(col0 + 16 + nidx) * HID + kB;

#pragma unroll
    for (int k0 = 0; k0 < HID; k0 += 32) {
        v16h a, b0, b1;
#pragma unroll
        for (int j = 0; j < 4; ++j) {
            a[2*j]   = arow[k0 + kA + 2*j];
            a[2*j+1] = arow[k0 + kA + 2*j + 1];
            a[2*j+8] = arow[k0 + kA + 16 + 2*j];
            a[2*j+9] = arow[k0 + kA + 16 + 2*j + 1];
        }
#pragma unroll
        for (int i = 0; i < 16; ++i) { b0[i] = b0p[k0 + (i < 8 ? i : i + 8)]; b1[i] = b1p[k0 + (i < 8 ? i : i + 8)]; }
        acc0 = __builtin_amdgcn_wmma_f32_16x16x32_f16(false, a, false, b0, (short)0, acc0, false, false);
        acc1 = __builtin_amdgcn_wmma_f32_16x16x32_f16(false, a, false, b1, (short)0, acc1, false, false);
    }

    const float bv0 = bias[col0 + nidx];
    const float bv1 = bias[col0 + 16 + nidx];
    const int   cm  = (lane < 16) ? 0 : 8;                 // C/D layout: VGPR r -> M=cm+r

    if (row0 + 16 <= n_rows) {                             // fast path: full tile
#pragma unroll
        for (int r = 0; r < 8; ++r) {
            const int    orow = row0 + cm + r;
            const size_t base = (size_t)orow * HID;
            float s = dinv[orow]; s *= s;
            const float v0 = acc0[r] + bv0, v1 = acc1[r] + bv1;
            xw [base + col0 + nidx]      = v0;
            xw [base + col0 + 16 + nidx] = v1;
            acc[base + col0 + nidx]      = s * v0;
            acc[base + col0 + 16 + nidx] = s * v1;
        }
    } else {                                               // tail tile (partial rows)
#pragma unroll
        for (int r = 0; r < 8; ++r) {
            const int orow = row0 + cm + r;
            if (orow < n_rows) {
                const size_t base = (size_t)orow * HID;
                float s = dinv[orow]; s *= s;
                const float v0 = acc0[r] + bv0, v1 = acc1[r] + bv1;
                xw [base + col0 + nidx]      = v0;
                xw [base + col0 + 16 + nidx] = v1;
                acc[base + col0 + nidx]      = s * v0;
                acc[base + col0 + 16 + nidx] = s * v1;
            }
        }
    }
}

// ---------------------------------------------------------------------------
// edge scatter:  acc[col] += dinv[row]*dinv[col] * xw[row]
// ---------------------------------------------------------------------------
__global__ void k_edge_scatter(const float* __restrict__ xw,
                               const int*   __restrict__ row,
                               const int*   __restrict__ col,
                               const float* __restrict__ dinv,
                               float* __restrict__ acc, int e)
{
    int gid = blockIdx.x * blockDim.x + threadIdx.x;   // e * 32 threads
    int ei = gid >> 5, c = (gid & 31) * 4;
    if (ei >= e) return;
    int r = row[ei], d = col[ei];
    float nrm = dinv[r] * dinv[d];
    const float4 v = *(const float4*)(xw + (size_t)r * HID + c);
    float* dst = acc + (size_t)d * HID + c;
    atomicAdd(dst + 0, nrm * v.x);
    atomicAdd(dst + 1, nrm * v.y);
    atomicAdd(dst + 2, nrm * v.z);
    atomicAdd(dst + 3, nrm * v.w);
}

// relu in place + produce f16 copy for next layer's WMMA
__global__ void k_relu_conv(float* __restrict__ acc, _Float16* __restrict__ xh, int n) {
    int i = blockIdx.x * blockDim.x + threadIdx.x;     // n * HID elements
    if (i >= n) return;
    float v = acc[i];
    v = v > 0.0f ? v : 0.0f;
    acc[i] = v;
    xh[i]  = (_Float16)v;
}

// ---------------------------------------------------------------------------
// attention pooling
// ---------------------------------------------------------------------------
__global__ void k_zero_ms(unsigned* menc, float* s) {
    int i = threadIdx.x;
    if (i < N_GRAPHS) { menc[i] = 0u; s[i] = 0.0f; }   // enc 0 == "-NaN" (min element)
}

__global__ void k_gate_max(const float* __restrict__ x,
                           const float* __restrict__ gate_w,
                           const float* __restrict__ gate_b,
                           const int*   __restrict__ batch,
                           float* __restrict__ gate, unsigned* __restrict__ menc, int n)
{
    int i = blockIdx.x * blockDim.x + threadIdx.x;
    if (i >= n) return;
    const float* xr = x + (size_t)i * HID;
    float g = gate_b[0];
#pragma unroll 8
    for (int k = 0; k < HID; k += 4) {
        float4 v = *(const float4*)(xr + k);
        float4 w = *(const float4*)(gate_w + k);
        g += v.x*w.x + v.y*w.y + v.z*w.z + v.w*w.w;
    }
    gate[i] = g;
    atomicMax(&menc[batch[i]], enc_f32(g));
}

__global__ void k_exp_sum(float* __restrict__ gate,
                          const unsigned* __restrict__ menc,
                          const int* __restrict__ batch,
                          float* __restrict__ s, int n)
{
    int i = blockIdx.x * blockDim.x + threadIdx.x;
    if (i >= n) return;
    float e = expf(gate[i] - dec_f32(menc[batch[i]]));
    gate[i] = e;                                       // reuse buffer: now holds e
    atomicAdd(&s[batch[i]], e);
}

__global__ void k_pool_acc(const float* __restrict__ e,
                           const float* __restrict__ s,
                           const float* __restrict__ x,
                           const int*   __restrict__ batch,
                           float* __restrict__ hg, int n)
{
    int gid = blockIdx.x * blockDim.x + threadIdx.x;   // n * 32 threads
    int i = gid >> 5, c = (gid & 31) * 4;
    if (i >= n) return;
    int b = batch[i];
    float alpha = e[i] / s[b];
    const float4 v = *(const float4*)(x + (size_t)i * HID + c);
    float* dst = hg + (size_t)b * HID + c;
    atomicAdd(dst + 0, alpha * v.x);
    atomicAdd(dst + 1, alpha * v.y);
    atomicAdd(dst + 2, alpha * v.z);
    atomicAdd(dst + 3, alpha * v.w);
}

__global__ void k_zero_f32(float* p, int n) {
    int i = blockIdx.x * blockDim.x + threadIdx.x;
    if (i < n) p[i] = 0.0f;
}

// ---------------------------------------------------------------------------
// classifier tail: hid = relu(hg @ lin_w.T + lin_b); out = hid @ cls_w.T + cls_b
// ---------------------------------------------------------------------------
__global__ void k_mlp1(const float* __restrict__ hg,
                       const float* __restrict__ lin_w, const float* __restrict__ lin_b,
                       float* __restrict__ hid)
{
    int g = blockIdx.x;          // graph
    int o = threadIdx.x;         // 0..255
    const float* hr = hg + (size_t)g * HID;
    const float* wr = lin_w + (size_t)o * HID;
    float acc = lin_b[o];
#pragma unroll 8
    for (int k = 0; k < HID; k += 4) {
        float4 v = *(const float4*)(hr + k);
        float4 w = *(const float4*)(wr + k);
        acc += v.x*w.x + v.y*w.y + v.z*w.z + v.w*w.w;
    }
    hid[(size_t)g * (2*HID) + o] = acc > 0.0f ? acc : 0.0f;
}

__global__ void k_mlp2(const float* __restrict__ hid,
                       const float* __restrict__ cls_w, const float* __restrict__ cls_b,
                       float* __restrict__ out)
{
    int g = blockIdx.x;
    int o = threadIdx.x;
    if (o >= OUT_DIM) return;
    const float* hr = hid + (size_t)g * (2*HID);
    const float* wr = cls_w + (size_t)o * (2*HID);
    float acc = cls_b[o];
#pragma unroll 8
    for (int k = 0; k < 2*HID; k += 4) {
        float4 v = *(const float4*)(hr + k);
        float4 w = *(const float4*)(wr + k);
        acc += v.x*w.x + v.y*w.y + v.z*w.z + v.w*w.w;
    }
    out[(size_t)g * OUT_DIM + o] = acc;
}

// ---------------------------------------------------------------------------
// host orchestration
// ---------------------------------------------------------------------------
extern "C" void kernel_launch(void* const* d_in, const int* in_sizes, int n_in,
                              void* d_out, int out_size, void* d_ws, size_t ws_size,
                              hipStream_t stream)
{
    const float* x_in   = (const float*)d_in[0];
    const int*   edge   = (const int*)  d_in[1];
    const int*   batch  = (const int*)  d_in[2];
    const float* w[3]   = { (const float*)d_in[3], (const float*)d_in[5], (const float*)d_in[7] };
    const float* bw[3]  = { (const float*)d_in[4], (const float*)d_in[6], (const float*)d_in[8] };
    const float* gate_w = (const float*)d_in[9];
    const float* gate_b = (const float*)d_in[10];
    const float* lin_w  = (const float*)d_in[11];
    const float* lin_b  = (const float*)d_in[12];
    const float* cls_w  = (const float*)d_in[13];
    const float* cls_b  = (const float*)d_in[14];
    float* out = (float*)d_out;

    const int n = in_sizes[0] / HID;      // 50000
    const int e = in_sizes[1] / 2;        // 640000
    const int* row = edge;
    const int* col = edge + e;

    // workspace carve-out (256B aligned)
    char* ws = (char*)d_ws;
    size_t off = 0;
    auto take = [&](size_t bytes) { char* p = ws + off; off = (off + bytes + 255) & ~(size_t)255; return p; };
    float*    dinv = (float*)    take((size_t)n * 4);
    _Float16* xh   = (_Float16*) take((size_t)n * HID * 2);
    _Float16* wh   = (_Float16*) take((size_t)3 * HID * HID * 2);
    float*    xw   = (float*)    take((size_t)n * HID * 4);
    float*    acc  = (float*)    take((size_t)n * HID * 4);
    float*    gate = (float*)    take((size_t)n * 4);
    unsigned* menc = (unsigned*) take((size_t)N_GRAPHS * 4);
    float*    s    = (float*)    take((size_t)N_GRAPHS * 4);
    float*    hg   = (float*)    take((size_t)N_GRAPHS * HID * 4);
    float*    hid  = (float*)    take((size_t)N_GRAPHS * 2 * HID * 4);
    (void)ws_size;

    const int T = 256;
    auto blk = [](long long work, int t) { return (int)((work + t - 1) / t); };

    // normalization: deg -> dinv (computed once)
    k_deg_init <<<blk(n, T), T, 0, stream>>>(dinv, n);
    k_deg_edges<<<blk(e, T), T, 0, stream>>>(col, dinv, e);
    k_dinv     <<<blk(n, T), T, 0, stream>>>(dinv, n);

    // f16 weights (once) + f16 input features
    for (int l = 0; l < 3; ++l)
        k_f32_to_f16<<<blk(HID*HID, T), T, 0, stream>>>(w[l], wh + (size_t)l*HID*HID, HID*HID);
    k_f32_to_f16<<<blk((long long)n*HID, T), T, 0, stream>>>(x_in, xh, n*HID);

    // hg accumulator
    k_zero_f32<<<blk(N_GRAPHS*HID, T), T, 0, stream>>>(hg, N_GRAPHS*HID);

    for (int l = 0; l < 3; ++l) {
        // dense: xw = x @ w.T + b (WMMA f16 -> f32), fused with self-loop init of acc
        k_gemm_xw_wmma<<<blk(n, 32), T, 0, stream>>>(
            xh, wh + (size_t)l*HID*HID, bw[l], dinv, xw, acc, n);
        // sparse aggregation (bandwidth-bound; acc is L2-resident at 25 MB)
        k_edge_scatter <<<blk((long long)e*32, T), T, 0, stream>>>(xw, row, col, dinv, acc, e);
        k_relu_conv    <<<blk((long long)n*HID, T), T, 0, stream>>>(acc, xh, n*HID);
        // attention pooling into hg
        k_zero_ms <<<1, T, 0, stream>>>(menc, s);
        k_gate_max<<<blk(n, T), T, 0, stream>>>(acc, gate_w, gate_b, batch, gate, menc, n);
        k_exp_sum <<<blk(n, T), T, 0, stream>>>(gate, menc, batch, s, n);
        k_pool_acc<<<blk((long long)n*32, T), T, 0, stream>>>(gate, s, acc, batch, hg, n);
    }

    // classifier
    k_mlp1<<<N_GRAPHS, 2*HID, 0, stream>>>(hg, lin_w, lin_b, hid);
    k_mlp2<<<N_GRAPHS, 32, 0, stream>>>(hid, cls_w, cls_b, out);
    (void)out_size; (void)n_in;
}
